// CausalSelfAttention_20160576487897
// MI455X (gfx1250) — compile-verified, matrix operand order checked by reading, not run on hardware
//
#include <hip/hip_runtime.h>
#include <hip/hip_bf16.h>

typedef __attribute__((ext_vector_type(16))) __bf16 v16bf;
typedef __attribute__((ext_vector_type(8)))  __bf16 v8bf;
typedef __attribute__((ext_vector_type(8)))  float  v8f;
typedef __attribute__((ext_vector_type(4)))  unsigned v4u;
typedef __attribute__((ext_vector_type(8)))  int    v8i;
typedef __attribute__((ext_vector_type(4)))  int    v4i;

#if __has_builtin(__builtin_amdgcn_tensor_load_to_lds)
#define HAVE_TDM 1
#else
#define HAVE_TDM 0
#endif

namespace {
constexpr int H = 64, HK = 8, HD = 64, DM = 1024, S = 1024, WINDOW = 128;
constexpr int QKD = H * HD;    // 4096
constexpr int KVD = HK * HD;   // 512
constexpr float SCALE = 0.125f; // 1/sqrt(64)

// ISA 16-bit A/B operand tile: per lane, 16 bf16 covering k = {0..7} then
// {16..23} relative to (k0 + 8*half). Both runs are contiguous -> two 16B loads.
__device__ __forceinline__ v16bf load_tile(const __bf16* p) {
  v8bf lo = *(const v8bf*)p;
  v8bf hi = *(const v8bf*)(p + 16);
  return __builtin_shufflevector(lo, hi, 0, 1, 2, 3, 4, 5, 6, 7,
                                 8, 9, 10, 11, 12, 13, 14, 15);
}
} // namespace

// One-time fp32 -> bf16 conversion (8 elements/thread). n % 2048 == 0.
__global__ __launch_bounds__(256) void cvt_f32_bf16(
    const float* __restrict__ in, __bf16* __restrict__ out) {
  const size_t i = ((size_t)blockIdx.x * blockDim.x + threadIdx.x) * 8;
  float4 f0 = *(const float4*)(in + i);
  float4 f1 = *(const float4*)(in + i + 4);
  v8bf o;
  o[0] = (__bf16)f0.x; o[1] = (__bf16)f0.y; o[2] = (__bf16)f0.z; o[3] = (__bf16)f0.w;
  o[4] = (__bf16)f1.x; o[5] = (__bf16)f1.y; o[6] = (__bf16)f1.z; o[7] = (__bf16)f1.w;
  *(v8bf*)(out + i) = o;
}

// out = A[M,K] @ W[N,K]^T + bias[N]; A,W bf16.
// Block (8 waves) owns a 256x64 tile; the 64x32 weight slab per K-step is
// staged into LDS by the Tensor Data Mover (double-buffered, TENSORcnt).
// MODE 0: fp32 out [M,N]; 1: bf16 out [M,N]; 2: bf16 out transposed [N,M].
template <int MODE>
__global__ __launch_bounds__(256) void gemm_bias_bf16(
    const __bf16* __restrict__ A, const __bf16* __restrict__ W,
    const float* __restrict__ bias, void* __restrict__ outp,
    int M, int N, int K) {
  __shared__ alignas(16) __bf16 wslab[2][64 * 32]; // [buf][n(64) x k(32)]
  const int lane = (int)(threadIdx.x & 31);
  const int wib  = (int)(threadIdx.x >> 5);
  const int nbx  = N >> 6;
  const int bm   = (int)blockIdx.x / nbx;
  const int bn   = (int)blockIdx.x % nbx;
  const int n0   = bn << 6;
  const int m0   = (bm << 8) + (wib << 5);
  const int row = lane & 15, half = lane >> 4;
  const int nsteps = K >> 5;

#if HAVE_TDM
  auto issue_tdm = [&](int buf, int k0) {
    const unsigned long long gaddr = (unsigned long long)(void*)W +
        ((unsigned long long)(unsigned)n0 * (unsigned)K + (unsigned)k0) * 2ull;
    const unsigned lds = (unsigned)(unsigned long long)(void*)&wslab[buf][0];
    v4u g0;
    g0[0] = 1u;                                   // count=1, user mode
    g0[1] = lds;                                  // lds_addr (bytes)
    g0[2] = (unsigned)gaddr;                      // global_addr[31:0]
    g0[3] = (unsigned)((gaddr >> 32) & 0x01FFFFFFull) | 0x80000000u; // [56:32] | type=2
    v8i g1;
    g1[0] = 0x00010000;                           // data_size = 2 bytes
    g1[1] = (int)(((unsigned)K & 0xFFFFu) << 16); // tensor_dim0[15:0] @ bits 63:48
    g1[2] = (int)((((unsigned)K >> 16) & 0xFFFFu) | (((unsigned)N & 0xFFFFu) << 16));
    g1[3] = (int)((((unsigned)N >> 16) & 0xFFFFu) | (32u << 16)); // tile_dim0=32
    g1[4] = 64;                                   // tile_dim1=64, tile_dim2=0
    g1[5] = K;                                    // tensor_dim0_stride[31:0]
    g1[6] = 0;
    g1[7] = 0;
    v4i gz = {0, 0, 0, 0};
#if defined(__clang_major__) && (__clang_major__ >= 23)
    v8i gz8 = {0, 0, 0, 0, 0, 0, 0, 0};
    __builtin_amdgcn_tensor_load_to_lds(g0, g1, gz, gz, gz8, 0);
#else
    __builtin_amdgcn_tensor_load_to_lds(g0, g1, gz, gz, 0);
#endif
  };
#else
  auto stage_slab = [&](int buf, int k0) { // cooperative fallback: 256 thr x 8 elem
    const int idx = (int)threadIdx.x * 8;
    const int n = idx >> 5, k = idx & 31;
    *(v8bf*)&wslab[buf][idx] = *(const v8bf*)(W + (size_t)(n0 + n) * K + k0 + k);
  };
#endif

  v8f acc[2][4] = {};
  const __bf16* a0p = A + (size_t)(m0 + row) * K + half * 8;
  const __bf16* a1p = a0p + (size_t)16 * K;

#if HAVE_TDM
  if (wib == 0) {
    issue_tdm(0, 0);
    __builtin_amdgcn_s_wait_tensorcnt(0);
  }
  __syncthreads();
#endif

  for (int i = 0; i < nsteps; ++i) {
    const int k0 = i << 5;
#if HAVE_TDM
    if (wib == 0 && i + 1 < nsteps) issue_tdm((i + 1) & 1, k0 + 32);
    const __bf16* ws = &wslab[i & 1][0];
#else
    stage_slab(0, k0);
    __syncthreads();
    const __bf16* ws = &wslab[0][0];
#endif
    v16bf a0 = load_tile(a0p + k0);
    v16bf a1 = load_tile(a1p + k0);
#pragma unroll
    for (int j = 0; j < 4; ++j) {
      v16bf bm = load_tile(ws + (j * 16 + row) * 32 + 8 * half);
      acc[0][j] = __builtin_amdgcn_wmma_f32_16x16x32_bf16(
          false, a0, false, bm, (short)0, acc[0][j], false, false);
      acc[1][j] = __builtin_amdgcn_wmma_f32_16x16x32_bf16(
          false, a1, false, bm, (short)0, acc[1][j], false, false);
    }
#if HAVE_TDM
    if (wib == 0) __builtin_amdgcn_s_wait_tensorcnt(0);
#endif
    __syncthreads();
  }

#pragma unroll
  for (int i = 0; i < 2; ++i) {
#pragma unroll
    for (int j = 0; j < 4; ++j) {
      const int n = n0 + j * 16 + row;
      const float bb = bias[n];
#pragma unroll
      for (int v = 0; v < 8; ++v) {
        const int m = m0 + i * 16 + v + 8 * half;
        const float val = acc[i][j][v] + bb;
        if (MODE == 0) {
          ((float*)outp)[(size_t)m * N + n] = val;
        } else if (MODE == 1) {
          ((__bf16*)outp)[(size_t)m * N + n] = (__bf16)val;
        } else {
          ((__bf16*)outp)[(size_t)n * M + m] = (__bf16)val;
        }
      }
    }
  }
}

// Sliding-window causal attention with per-head sink logit.
// q,k bf16 row-major; vT bf16 [HK*HD, B*S]; o bf16 row-major.
// One wave per (b, h, 16-query tile).
__global__ __launch_bounds__(256) void attn_swa_sink(
    const __bf16* __restrict__ q, const __bf16* __restrict__ kbuf,
    const __bf16* __restrict__ vT, const float* __restrict__ sinks,
    __bf16* __restrict__ o, int nq, int Mtot) {
  __shared__ alignas(16) __bf16 pbuf[8][16 * 32];
  const int wid  = (int)((blockIdx.x * blockDim.x + threadIdx.x) >> 5);
  const int lane = (int)(threadIdx.x & 31);
  const int wib  = (int)(threadIdx.x >> 5);
  const int b  = wid / (H * nq);
  const int r  = wid % (H * nq);
  const int h  = r / nq, qt = r % nq;
  const int hk = h / (H / HK);
  const int row = lane & 15, half = lane >> 4;

  const __bf16* qrow = q + (size_t)(b * S + qt * 16 + row) * QKD + h * HD + half * 8;
  const v16bf qa0 = load_tile(qrow);
  const v16bf qa1 = load_tile(qrow + 32);

  float rmax[8], rsum[8];
#pragma unroll
  for (int v = 0; v < 8; ++v) { rmax[v] = -1e30f; rsum[v] = 0.f; }
  v8f oacc[4] = {};

  int qlo = qt * 16 - (WINDOW - 1); if (qlo < 0) qlo = 0;
  const int kb_lo = qlo >> 5;
  const int kb_hi = (qt * 16 + 15) >> 5;

  for (int kb = kb_lo; kb <= kb_hi; ++kb) {
    const int ks = kb << 5;
    v8f st[2];
#pragma unroll
    for (int t = 0; t < 2; ++t) {
      const __bf16* krow =
          kbuf + (size_t)(b * S + ks + t * 16 + row) * KVD + hk * HD + half * 8;
      v8f z = {};
      z = __builtin_amdgcn_wmma_f32_16x16x32_bf16(
          false, qa0, false, load_tile(krow), (short)0, z, false, false);
      z = __builtin_amdgcn_wmma_f32_16x16x32_bf16(
          false, qa1, false, load_tile(krow + 32), (short)0, z, false, false);
      st[t] = z;
    }
#pragma unroll
    for (int vv = 0; vv < 8; ++vv) {
      const int qpos = qt * 16 + vv + 8 * half;
      const int kp0 = ks + row, kp1 = ks + 16 + row;
      const bool a0 = (qpos >= kp0) && (qpos - kp0 < WINDOW);
      const bool a1 = (qpos >= kp1) && (qpos - kp1 < WINDOW);
      const float s0 = st[0][vv] * SCALE, s1 = st[1][vv] * SCALE;
      float tm = fmaxf(a0 ? s0 : -1e30f, a1 ? s1 : -1e30f);
#pragma unroll
      for (int off = 1; off < 16; off <<= 1) tm = fmaxf(tm, __shfl_xor(tm, off, 32));
      const float mnew = fmaxf(rmax[vv], tm);
      const float corr = __expf(rmax[vv] - mnew);
      const float p0 = a0 ? __expf(s0 - mnew) : 0.f;
      const float p1 = a1 ? __expf(s1 - mnew) : 0.f;
      float ps = p0 + p1;
#pragma unroll
      for (int off = 1; off < 16; off <<= 1) ps += __shfl_xor(ps, off, 32);
      rsum[vv] = rsum[vv] * corr + ps;
      rmax[vv] = mnew;
#pragma unroll
      for (int j = 0; j < 4; ++j) oacc[j][vv] *= corr;
      const int prow = vv + 8 * half;
      pbuf[wib][prow * 32 + row]      = (__bf16)p0;
      pbuf[wib][prow * 32 + 16 + row] = (__bf16)p1;
    }
    asm volatile("s_wait_dscnt 0" ::: "memory"); // wave-local LDS RAW fence
    const v16bf pa = load_tile(&pbuf[wib][row * 32 + 8 * half]);
#pragma unroll
    for (int j = 0; j < 4; ++j) {
      const __bf16* vp =
          vT + (size_t)(hk * HD + j * 16 + row) * Mtot + b * S + ks + half * 8;
      oacc[j] = __builtin_amdgcn_wmma_f32_16x16x32_bf16(
          false, pa, false, load_tile(vp), (short)0, oacc[j], false, false);
    }
  }

  const float slog = sinks[h];
#pragma unroll
  for (int vv = 0; vv < 8; ++vv) {
    const float mf = fmaxf(rmax[vv], slog);
    const float z  = rsum[vv] * __expf(rmax[vv] - mf) + __expf(slog - mf);
    const float rr = __expf(rmax[vv] - mf) / z;
    const int m = qt * 16 + vv + 8 * half;
#pragma unroll
    for (int j = 0; j < 4; ++j) {
      o[(size_t)(b * S + m) * QKD + h * HD + j * 16 + row] =
          (__bf16)(oacc[j][vv] * rr);
    }
  }
}

extern "C" void kernel_launch(void* const* d_in, const int* in_sizes, int n_in,
                              void* d_out, int out_size, void* d_ws, size_t ws_size,
                              hipStream_t stream) {
  (void)n_in; (void)out_size; (void)ws_size;
  const float* x     = (const float*)d_in[0];
  const float* Wq    = (const float*)d_in[1];
  const float* bq    = (const float*)d_in[2];
  const float* Wk    = (const float*)d_in[3];
  const float* bk    = (const float*)d_in[4];
  const float* Wv    = (const float*)d_in[5];
  const float* bv    = (const float*)d_in[6];
  const float* Wo    = (const float*)d_in[7];
  const float* bo    = (const float*)d_in[8];
  const float* sinks = (const float*)d_in[9];
  float* out = (float*)d_out;

  const int B = in_sizes[0] / (S * DM);
  const int M = B * S;

  __bf16* p = (__bf16*)d_ws;
  __bf16* xb  = p; p += (size_t)M * DM;
  __bf16* Wqb = p; p += (size_t)QKD * DM;
  __bf16* Wkb = p; p += (size_t)KVD * DM;
  __bf16* Wvb = p; p += (size_t)KVD * DM;
  __bf16* Wob = p; p += (size_t)DM * QKD;
  __bf16* qb  = p; p += (size_t)M * QKD;
  __bf16* kb  = p; p += (size_t)M * KVD;
  __bf16* vTb = p; p += (size_t)KVD * M;   // transposed V [512, M]
  __bf16* ob  = p; p += (size_t)M * QKD;

  auto cvt = [&](const float* in, __bf16* o2, size_t n) {
    cvt_f32_bf16<<<(unsigned)(n / 2048), 256, 0, stream>>>(in, o2);
  };
  cvt(x,  xb,  (size_t)M * DM);
  cvt(Wq, Wqb, (size_t)QKD * DM);
  cvt(Wk, Wkb, (size_t)KVD * DM);
  cvt(Wv, Wvb, (size_t)KVD * DM);
  cvt(Wo, Wob, (size_t)DM * QKD);

  const int nblkQ = (M / 256) * (QKD / 64);
  const int nblkK = (M / 256) * (KVD / 64);
  const int nblkO = (M / 256) * (DM / 64);
  gemm_bias_bf16<1><<<nblkQ, 256, 0, stream>>>(xb, Wqb, bq, qb,  M, QKD, DM);
  gemm_bias_bf16<1><<<nblkK, 256, 0, stream>>>(xb, Wkb, bk, kb,  M, KVD, DM);
  gemm_bias_bf16<2><<<nblkK, 256, 0, stream>>>(xb, Wvb, bv, vTb, M, KVD, DM);

  {
    const int nq = S / 16;
    const int waves = B * H * nq;
    attn_swa_sink<<<waves * 32 / 256, 256, 0, stream>>>(qb, kb, vTb, sinks, ob, nq, M);
  }

  gemm_bias_bf16<0><<<nblkO, 256, 0, stream>>>(ob, Wob, bo, out, M, DM, QKD);
}
